// ResBlock_W1A1_81458349735955
// MI455X (gfx1250) — compile-verified, matrix operand order checked by reading, not run on hardware
//
#include <hip/hip_runtime.h>

// MI455X / gfx1250, wave32. Binary convs -> V_WMMA_I32_16X16X64_IU8 (exact for
// +/-1 data); fp32 skip conv -> V_WMMA_F32_16X16X4_F32 fused into conv2 epilogue.
// Staging uses GLOBAL_LOAD_ASYNC_TO_LDS_B128 (ASYNCcnt) on the CDNA5 async path.

typedef int   v8i __attribute__((ext_vector_type(8)));
typedef float v8f __attribute__((ext_vector_type(8)));
typedef float v2f __attribute__((ext_vector_type(2)));

#define BN_EPS 1e-5f
#define B_   16
#define CIN  128
#define COUT 256
#define H_   64
#define W_   64
#define T_   512

// workspace layout (bytes)
#define OFF_A1   0u          // 16*64*64*128 int8  =  8,388,608
#define OFF_A2   8388608u    // 16*64*64*256 int8  = 16,777,216
#define OFF_B1   25165824u   // 18 kchunks *16 nfrags *32 lanes *32B = 294,912
#define OFF_B2   25460736u   // 36 kchunks *16*32*32 = 589,824
#define OFF_ADD1 26050560u   // 16*256 f32
#define OFF_INV2 26066944u   // 256 f32
#define OFF_SH2  26067968u   // 256 f32

__device__ __forceinline__ void async_b128(unsigned lds_off, unsigned long long gaddr) {
  asm volatile("global_load_async_to_lds_b128 %0, %1, off"
               :: "v"(lds_off), "v"(gaddr) : "memory");
}
__device__ __forceinline__ void wait_async0() {
  asm volatile("s_wait_asynccnt 0x0" ::: "memory");
}

// ---------------------------------------------------------------------------
// Pack binarized conv weights into WMMA-B (64x16 int8) fragment-ready layout:
// bp[((kc*16+nf)*32+lane)*8 + v], kc = tap*CH + chunk, dword d of a column:
// d = (v&1) | (lg<<1) | ((v>>1)<<2), lane = n_local + 16*lg.
__global__ void pack_w_kernel(const float* __restrict__ w, int Cin, int CH,
                              int* __restrict__ bp, int total) {
  int idx = blockIdx.x * 256 + threadIdx.x;
  if (idx >= total) return;
  int v    = idx & 7;
  int lane = (idx >> 3) & 31;
  int nf   = (idx >> 8) & 15;
  int kc   = idx >> 12;
  int lg = lane >> 4, nl = lane & 15;
  int d  = (v & 1) | (lg << 1) | ((v >> 1) << 2);
  int n  = nf * 16 + nl;
  int tap = kc / CH, chunk = kc % CH;
  int ky = tap / 3, kx = tap % 3;
  unsigned dw = 0;
  #pragma unroll
  for (int j = 0; j < 4; j++) {
    int cin = chunk * 64 + 4 * d + j;
    float wv = w[((n * Cin + cin) * 3 + ky) * 3 + kx];
    unsigned byte = (wv >= 0.f) ? 0x01u : 0xFFu;  // +1 / -1 as int8
    dw |= byte << (8 * j);
  }
  bp[idx] = (int)dw;
}

// add1[b][n] = b1[n] + bt[n] + dot(t[b], wt[n]); also fold BN2 into scale/shift
__global__ void temb_kernel(const float* __restrict__ t,  const float* __restrict__ wt,
                            const float* __restrict__ bt, const float* __restrict__ b1,
                            const float* __restrict__ g2, const float* __restrict__ be2,
                            const float* __restrict__ m2, const float* __restrict__ va2,
                            float* __restrict__ add1, float* __restrict__ inv2,
                            float* __restrict__ sh2) {
  int tid = blockIdx.x * 256 + threadIdx.x;  // 16*256
  int b = tid >> 8, n = tid & 255;
  float s = b1[n] + bt[n];
  for (int k = 0; k < T_; k++) s += t[b * T_ + k] * wt[n * T_ + k];
  add1[tid] = s;
  if (tid < COUT) {
    float iv = g2[tid] * rsqrtf(va2[tid] + BN_EPS);
    inv2[tid] = iv;
    sh2[tid]  = be2[tid] - m2[tid] * iv;
  }
}

// BN1 + binarize x -> NHWC int8 (C contiguous: GEMM-K-contiguous)
__global__ void binx_kernel(const float* __restrict__ x, const float* __restrict__ g1,
                            const float* __restrict__ be1, const float* __restrict__ m1,
                            const float* __restrict__ v1, char* __restrict__ a1) {
  int idx = blockIdx.x * 256 + threadIdx.x;  // 16*128*64*16
  int xg = idx & 15;
  int y  = (idx >> 4) & 63;
  int c  = (idx >> 10) & 127;
  int b  = idx >> 17;
  float iv = g1[c] * rsqrtf(v1[c] + BN_EPS);
  float sh = be1[c] - m1[c] * iv;
  float4 xv = *(const float4*)(x + (((long)(b * CIN + c) * H_ + y) * W_ + xg * 4));
  float vv[4] = {xv.x, xv.y, xv.z, xv.w};
  #pragma unroll
  for (int i = 0; i < 4; i++) {
    float val = vv[i] * iv + sh;
    a1[((long)((b * H_ + y) * W_ + xg * 4 + i)) * CIN + c] = (val >= 0.f) ? (char)1 : (char)-1;
  }
}

// conv1 (binary, K=1152) fused with +b1 +temb, BN2, binarize -> a2 int8 NHWC.
// One block per image row (b,y); 8 waves as 2(M)x4(N): wave tile 32px x 64 couts.
__global__ __launch_bounds__(256) void conv1_kernel(
    const char* __restrict__ a1, const int* __restrict__ bp1,
    const float* __restrict__ add1, const float* __restrict__ inv2,
    const float* __restrict__ sh2, char* __restrict__ a2) {
  extern __shared__ char smem[];  // 3*66*128 bytes: 3 halo rows, zero-padded
  int bid = blockIdx.x;
  int b = bid >> 6, y = bid & 63;
  int tid = threadIdx.x;
  int* sAd = (int*)smem;
  // zero-fill halo (row out of image; columns 0 and 65) with DS stores
  for (int i = tid; i < 3 * 66 * 32; i += 256) {
    int r = i / (66 * 32), rem = i % (66 * 32);
    int col = rem >> 5;
    int row = y + r - 1, xc = col - 1;
    if (!(row >= 0 && row < H_ && xc >= 0 && xc < W_)) sAd[i] = 0;
  }
  // async-copy the valid interior straight into LDS (no VGPR round trip)
  #pragma unroll
  for (int r = 0; r < 3; r++) {
    int row = y + r - 1;
    if (row >= 0 && row < H_) {
      for (int j = tid; j < 512; j += 256) {  // 64 cols * 8 x 16B segments
        int col = j >> 3, seg = j & 7;
        unsigned dst = (unsigned)(unsigned long long)(smem + (r * 66 + col + 1) * CIN + seg * 16);
        unsigned long long src = (unsigned long long)(a1 +
            ((long)((b * H_ + row) * W_ + col)) * CIN + seg * 16);
        async_b128(dst, src);
      }
    }
  }
  wait_async0();
  __syncthreads();

  int lane = tid & 31, wv = tid >> 5;
  int mw = wv & 1, nw = wv >> 1;          // 2 x 4 wave grid
  int lg = lane >> 4, ml = lane & 15;
  v8i acc[2][4];
  v8i zero = {0, 0, 0, 0, 0, 0, 0, 0};
  #pragma unroll
  for (int t = 0; t < 2; t++)
    #pragma unroll
    for (int f = 0; f < 4; f++) acc[t][f] = zero;

  const int4* bb = (const int4*)bp1 + ((nw * 4) * 32 + lane) * 2;
  const char* ab0 = smem + (mw * 32 + ml) * CIN + 8 * lg;       // m-tile 0
  const char* ab1 = ab0 + 16 * CIN;                             // m-tile 1

  #pragma unroll
  for (int tap = 0; tap < 9; tap++) {
    #pragma unroll
    for (int chunk = 0; chunk < 2; chunk++) {
      const int kc = tap * 2 + chunk;
      const int aoff = ((tap / 3) * 66 + tap % 3) * CIN + chunk * 64;
      // B fragments first (global, clauseable), into distinct registers
      v8i bfr[4];
      #pragma unroll
      for (int f = 0; f < 4; f++) {
        const int4* p4 = bb + kc * 1024 + f * 64;
        int4 p = p4[0], q = p4[1];
        v8i bf = {p.x, p.y, p.z, p.w, q.x, q.y, q.z, q.w};
        bfr[f] = bf;
      }
      // A fragments from LDS
      v8i afr[2];
      #pragma unroll
      for (int t = 0; t < 2; t++) {
        const char* ap = (t ? ab1 : ab0) + aoff;
        unsigned long long u0 = *(const unsigned long long*)(ap);
        unsigned long long u1 = *(const unsigned long long*)(ap + 16);
        unsigned long long u2 = *(const unsigned long long*)(ap + 32);
        unsigned long long u3 = *(const unsigned long long*)(ap + 48);
        v8i af = {(int)u0, (int)(u0 >> 32), (int)u1, (int)(u1 >> 32),
                  (int)u2, (int)(u2 >> 32), (int)u3, (int)(u3 >> 32)};
        afr[t] = af;
      }
      #pragma unroll
      for (int t = 0; t < 2; t++)
        #pragma unroll
        for (int f = 0; f < 4; f++)
          acc[t][f] = __builtin_amdgcn_wmma_i32_16x16x64_iu8(true, afr[t], true, bfr[f],
                                                             acc[t][f], false, false);
    }
  }
  #pragma unroll
  for (int t = 0; t < 2; t++)
    #pragma unroll
    for (int f = 0; f < 4; f++) {
      int n = nw * 64 + f * 16 + ml;
      float addv = add1[b * COUT + n];
      float iv = inv2[n], sv = sh2[n];
      #pragma unroll
      for (int r = 0; r < 8; r++) {
        int m = r + 8 * lg;
        float val = (float)acc[t][f][r] + addv;
        val = val * iv + sv;
        a2[((long)((b * H_ + y) * W_ + mw * 32 + t * 16 + m)) * COUT + n] =
            (val >= 0.f) ? (char)1 : (char)-1;
      }
    }
}

// conv2 (binary, K=2304) + b2 + fused fp32 1x1 skip (f32 WMMA) + bs -> out NCHW
__global__ __launch_bounds__(256) void conv2_kernel(
    const char* __restrict__ a2, const int* __restrict__ bp2,
    const float* __restrict__ x, const float* __restrict__ wsk,
    const float* __restrict__ b2, const float* __restrict__ bs,
    float* __restrict__ out) {
  extern __shared__ char smem[];  // 3*66*256 int8 + 128*64 f32
  int bid = blockIdx.x;
  int b = bid >> 6, y = bid & 63;
  int tid = threadIdx.x;
  int* sAd = (int*)smem;
  float* sX = (float*)(smem + 3 * 66 * 256);
  // zero halos
  for (int i = tid; i < 3 * 66 * 64; i += 256) {
    int r = i / (66 * 64), rem = i % (66 * 64);
    int col = rem >> 6;
    int row = y + r - 1, xc = col - 1;
    if (!(row >= 0 && row < H_ && xc >= 0 && xc < W_)) sAd[i] = 0;
  }
  // async interior: binarized activations
  #pragma unroll
  for (int r = 0; r < 3; r++) {
    int row = y + r - 1;
    if (row >= 0 && row < H_) {
      for (int j = tid; j < 1024; j += 256) {  // 64 cols * 16 x 16B segments
        int col = j >> 4, seg = j & 15;
        unsigned dst = (unsigned)(unsigned long long)(smem + (r * 66 + col + 1) * COUT + seg * 16);
        unsigned long long src = (unsigned long long)(a2 +
            ((long)((b * H_ + row) * W_ + col)) * COUT + seg * 16);
        async_b128(dst, src);
      }
    }
  }
  // async: fp32 x row for the skip GEMM (128 ch x 64 px)
  for (int i = tid; i < 2048; i += 256) {     // 128 ch * 16 x 16B segments
    int c = i >> 4, seg = i & 15;
    unsigned dst = (unsigned)(unsigned long long)((char*)sX + c * 256 + seg * 16);
    unsigned long long src = (unsigned long long)((const char*)x +
        ((long)((b * CIN + c) * H_ + y)) * W_ * 4 + seg * 16);
    async_b128(dst, src);
  }
  wait_async0();
  __syncthreads();

  int lane = tid & 31, wv = tid >> 5;
  int mw = wv & 1, nw = wv >> 1;
  int lg = lane >> 4, ml = lane & 15;
  v8i acc[2][4];
  v8i zero = {0, 0, 0, 0, 0, 0, 0, 0};
  #pragma unroll
  for (int t = 0; t < 2; t++)
    #pragma unroll
    for (int f = 0; f < 4; f++) acc[t][f] = zero;

  const int4* bb = (const int4*)bp2 + ((nw * 4) * 32 + lane) * 2;
  const char* ab0 = smem + (mw * 32 + ml) * COUT + 8 * lg;
  const char* ab1 = ab0 + 16 * COUT;

  #pragma unroll
  for (int tap = 0; tap < 9; tap++) {
    #pragma unroll
    for (int chunk = 0; chunk < 4; chunk++) {
      const int kc = tap * 4 + chunk;
      const int aoff = ((tap / 3) * 66 + tap % 3) * COUT + chunk * 64;
      v8i bfr[4];
      #pragma unroll
      for (int f = 0; f < 4; f++) {
        const int4* p4 = bb + kc * 1024 + f * 64;
        int4 p = p4[0], q = p4[1];
        v8i bf = {p.x, p.y, p.z, p.w, q.x, q.y, q.z, q.w};
        bfr[f] = bf;
      }
      v8i afr[2];
      #pragma unroll
      for (int t = 0; t < 2; t++) {
        const char* ap = (t ? ab1 : ab0) + aoff;
        unsigned long long u0 = *(const unsigned long long*)(ap);
        unsigned long long u1 = *(const unsigned long long*)(ap + 16);
        unsigned long long u2 = *(const unsigned long long*)(ap + 32);
        unsigned long long u3 = *(const unsigned long long*)(ap + 48);
        v8i af = {(int)u0, (int)(u0 >> 32), (int)u1, (int)(u1 >> 32),
                  (int)u2, (int)(u2 >> 32), (int)u3, (int)(u3 >> 32)};
        afr[t] = af;
      }
      #pragma unroll
      for (int t = 0; t < 2; t++)
        #pragma unroll
        for (int f = 0; f < 4; f++)
          acc[t][f] = __builtin_amdgcn_wmma_i32_16x16x64_iu8(true, afr[t], true, bfr[f],
                                                             acc[t][f], false, false);
    }
  }

  // epilogue: int acc -> f32 (+b2+bs), then chain fp32 skip GEMM (K=128) via WMMA
  v8f cacc[2][4];
  #pragma unroll
  for (int t = 0; t < 2; t++)
    #pragma unroll
    for (int f = 0; f < 4; f++) {
      int n = nw * 64 + f * 16 + ml;
      float bias = b2[n] + bs[n];
      #pragma unroll
      for (int r = 0; r < 8; r++) cacc[t][f][r] = (float)acc[t][f][r] + bias;
    }
  const float* wb = wsk + (nw * 64 + ml) * CIN + 2 * lg;  // per-wave B base
  #pragma unroll
  for (int ks = 0; ks < 32; ks++) {
    int k0 = 4 * ks;
    v2f af[2];
    #pragma unroll
    for (int t = 0; t < 2; t++) {
      int px = mw * 32 + t * 16 + ml;
      af[t][0] = sX[(k0 + 2 * lg) * W_ + px];
      af[t][1] = sX[(k0 + 1 + 2 * lg) * W_ + px];
    }
    v2f bf[4];
    #pragma unroll
    for (int f = 0; f < 4; f++) bf[f] = *(const v2f*)(wb + f * 16 * CIN + k0);
    #pragma unroll
    for (int t = 0; t < 2; t++)
      #pragma unroll
      for (int f = 0; f < 4; f++)
        cacc[t][f] = __builtin_amdgcn_wmma_f32_16x16x4_f32(false, af[t], false, bf[f],
                                                           (short)0, cacc[t][f], false, false);
  }
  // coalesced b128 stores: 8 consecutive floats per lane per fragment
  #pragma unroll
  for (int t = 0; t < 2; t++)
    #pragma unroll
    for (int f = 0; f < 4; f++) {
      int n = nw * 64 + f * 16 + ml;
      float* base = out + ((long)(b * COUT + n)) * (H_ * W_) + y * W_ + mw * 32 + t * 16 + 8 * lg;
      float4 lo = {cacc[t][f][0], cacc[t][f][1], cacc[t][f][2], cacc[t][f][3]};
      float4 hi = {cacc[t][f][4], cacc[t][f][5], cacc[t][f][6], cacc[t][f][7]};
      *(float4*)(base) = lo;
      *(float4*)(base + 4) = hi;
    }
}

extern "C" void kernel_launch(void* const* d_in, const int* in_sizes, int n_in,
                              void* d_out, int out_size, void* d_ws, size_t ws_size,
                              hipStream_t stream) {
  (void)in_sizes; (void)n_in; (void)out_size; (void)ws_size;
  const float* x   = (const float*)d_in[0];
  const float* t   = (const float*)d_in[1];
  const float* g1  = (const float*)d_in[2];
  const float* be1 = (const float*)d_in[3];
  const float* m1  = (const float*)d_in[4];
  const float* v1  = (const float*)d_in[5];
  const float* w1  = (const float*)d_in[6];
  const float* b1  = (const float*)d_in[7];
  const float* g2  = (const float*)d_in[8];
  const float* be2 = (const float*)d_in[9];
  const float* m2  = (const float*)d_in[10];
  const float* va2 = (const float*)d_in[11];
  const float* w2  = (const float*)d_in[12];
  const float* b2  = (const float*)d_in[13];
  const float* wt  = (const float*)d_in[14];
  const float* bt  = (const float*)d_in[15];
  const float* wsk = (const float*)d_in[16];
  const float* bs  = (const float*)d_in[17];
  char* base = (char*)d_ws;
  char*  a1   = base + OFF_A1;
  char*  a2   = base + OFF_A2;
  int*   bp1  = (int*)(base + OFF_B1);
  int*   bp2  = (int*)(base + OFF_B2);
  float* add1 = (float*)(base + OFF_ADD1);
  float* inv2 = (float*)(base + OFF_INV2);
  float* sh2  = (float*)(base + OFF_SH2);
  float* out  = (float*)d_out;

  pack_w_kernel<<<288, 256, 0, stream>>>(w1, 128, 2, bp1, 73728);
  pack_w_kernel<<<576, 256, 0, stream>>>(w2, 256, 4, bp2, 147456);
  temb_kernel<<<16, 256, 0, stream>>>(t, wt, bt, b1, g2, be2, m2, va2, add1, inv2, sh2);
  binx_kernel<<<8192, 256, 0, stream>>>(x, g1, be1, m1, v1, a1);
  conv1_kernel<<<1024, 256, 3 * 66 * 128, stream>>>(a1, bp1, add1, inv2, sh2, a2);
  conv2_kernel<<<1024, 256, 3 * 66 * 256 + 128 * 64 * 4, stream>>>(a2, bp2, x, wsk, b2, bs, out);
}
